// SimpleSequenceGenerator_11261404250245
// MI455X (gfx1250) — compile-verified, matrix operand order checked by reading, not run on hardware
//
#include <hip/hip_runtime.h>
#include <hip/hip_bf16.h>

// ---------------- problem constants ----------------
#define KBSZ    32
#define KBEAM   4
#define KBB     128            // BSZ*BEAM
#define KVOCAB  50257
#define KVP     50272          // vocab padded to multiple of 16 (3142 tiles)
#define KNT     3142           // valid 16-col tiles
#define KNTP    3144           // padded to multiple of 8 (one WG covers 8 tiles)
#define KD      1024
#define KL      66             // MAXLEN+2
#define KSLEN   65             // MAXLEN+1
#define KMAXLEN 64
#define TPAD    1
#define TEOS    2
#define KNEG    (-1e9f)

typedef __bf16 v16bf __attribute__((ext_vector_type(16)));
typedef float  v8f   __attribute__((ext_vector_type(8)));

#if __has_builtin(__builtin_amdgcn_global_load_async_to_lds_b128)
#define USE_ASYNC_LDS 1
typedef int v4i_gcc __attribute__((vector_size(16)));
typedef __attribute__((address_space(1))) v4i_gcc as1_v4i;
typedef __attribute__((address_space(3))) v4i_gcc as3_v4i;
__device__ __forceinline__ void wait_async0() {
#if __has_builtin(__builtin_amdgcn_s_wait_asynccnt)
  __builtin_amdgcn_s_wait_asynccnt(0);
#else
  asm volatile("s_wait_asynccnt 0x0" ::: "memory");
#endif
}
#endif

__device__ __forceinline__ unsigned short f32_to_bf16(float f) {
  unsigned int u = __float_as_uint(f);
  unsigned int r = (u + 0x7FFFu + ((u >> 16) & 1u)) >> 16;
  return (unsigned short)r;
}

// ---------------- one-time: pack W (fp32 [D][VOCAB]) -> bf16 WMMA-B fragments ----
// Bp[frag][lane][16 bf16], frag = n_tile*32 + k_tile (K in chunks of 32).
__global__ void pack_W_kernel(const float* __restrict__ W, unsigned short* __restrict__ Bp) {
  size_t idx = (size_t)blockIdx.x * blockDim.x + threadIdx.x;   // one (frag,lane)
  size_t nfl = (size_t)KNTP * 32 * 32;
  if (idx >= nfl) return;
  int lane = (int)(idx & 31);
  size_t frag = idx >> 5;
  int k_tile = (int)(frag & 31);
  int n_tile = (int)(frag >> 5);
  int lane16 = lane & 15;
  int half   = lane >> 4;
  int col = n_tile * 16 + lane16;
  unsigned short* out = Bp + frag * 512 + (size_t)lane * 16;
#pragma unroll
  for (int e = 0; e < 16; ++e) {
    int k = k_tile * 32 + (e >> 3) * 16 + half * 8 + (e & 7);
    float v = (col < KVOCAB) ? W[(size_t)k * KVOCAB + col] : 0.0f;
    out[e] = f32_to_bf16(v);
  }
}

// ---------------- init beam state ----------------
__global__ void init_state_kernel(int* tok, float* sc, float* fsc, int* ftok) {
  int i = blockIdx.x * blockDim.x + threadIdx.x;
  if (i < KBB * KL)          tok[i]  = ((i % KL) == 0) ? TEOS : TPAD;
  if (i < KBB * KSLEN)       sc[i]   = 0.0f;
  if (i < KBSZ * KBEAM)      fsc[i]  = KNEG;
  if (i < KBSZ * KBEAM * KL) ftok[i] = 0;
}

// ---------------- per step: h = emb[prev] + pos_emb[step], cvt bf16 ----------------
__global__ void build_h_kernel(const float* __restrict__ emb, const float* __restrict__ pos,
                               const int* __restrict__ tok, int step,
                               unsigned short* __restrict__ hA) {
  int row = blockIdx.x;                       // 0..127
  int prev = tok[row * KL + step];
  const float* e = emb + (size_t)prev * KD;
  const float* p = pos + (size_t)step * KD;
  for (int c = threadIdx.x; c < KD; c += blockDim.x)
    hA[(size_t)row * KD + c] = f32_to_bf16(e[c] + p[c]);
}

// ---------------- WMMA GEMM: logits[128][KVP] = h(128xK bf16) * W(KxN bf16) ------
// 128 threads = 4 waves; each wave owns TWO 16-col N-tiles (A fragments reused 2x),
// full M=128 (2x8 accumulator tiles).  K staged through double-buffered LDS in
// chunks of 64 (row stride 72 -> conflict-free banks), filled asynchronously
// (GLOBAL_LOAD_ASYNC_TO_LDS_B128 / ASYNCcnt) when the toolchain exposes it.
#define CH   64                 // K elements per chunk
#define CSTR 72                 // padded LDS row stride (elements)

__global__ void __launch_bounds__(128, 1)
gemm_logits_kernel(const unsigned short* __restrict__ hA,
                   const unsigned short* __restrict__ Bp,
                   float* __restrict__ logits) {
  __shared__ __align__(16) unsigned short sA[2][128 * CSTR];
  const int tid  = threadIdx.x;
  const int wave = tid >> 5;
  const int lane = tid & 31;
  const int lane16 = lane & 15;
  const int half   = lane >> 4;
  const int nt0 = blockIdx.x * 8 + wave * 2;      // < KNTP (Bp padded -> always safe)
  const int nt1 = nt0 + 1;

  v8f acc0[8], acc1[8];
#pragma unroll
  for (int m = 0; m < 8; ++m) {
    acc0[m] = (v8f){0.f,0.f,0.f,0.f,0.f,0.f,0.f,0.f};
    acc1[m] = (v8f){0.f,0.f,0.f,0.f,0.f,0.f,0.f,0.f};
  }

#ifdef USE_ASYNC_LDS
  // ---- async double-buffered staging ----
  // chunk = 128 rows x 64 bf16 = 1024 x 16B pieces; 8 pieces per thread.
  auto stage = [&](int buf, int kc) {
#pragma unroll
    for (int i = 0; i < 8; ++i) {
      int c   = i * 128 + tid;
      int row = c >> 3;
      int off = (c & 7) << 3;                     // element offset
      const unsigned short* g = hA + (size_t)row * KD + kc * CH + off;
      unsigned short* l = &sA[buf][row * CSTR + off];
      __builtin_amdgcn_global_load_async_to_lds_b128(
          (as1_v4i*)(__UINTPTR_TYPE__)g, (as3_v4i*)(__UINTPTR_TYPE__)l, 0, 0);
    }
  };
  stage(0, 0);
  wait_async0();
  __syncthreads();
#endif

  for (int kc = 0; kc < 16; ++kc) {               // 16 K-chunks of 64
#ifdef USE_ASYNC_LDS
    const int buf = kc & 1;
    if (kc < 15) stage(buf ^ 1, kc + 1);          // overlap next fill with compute
#else
    const int buf = 0;
    __syncthreads();
#pragma unroll
    for (int i = 0; i < 8; ++i) {                 // cooperative sync fill
      int c   = i * 128 + tid;
      int row = c >> 3;
      int off = (c & 7) << 3;
      uint4 v = *(const uint4*)(hA + (size_t)row * KD + kc * CH + off);
      *(uint4*)(&sA[buf][row * CSTR + off]) = v;
    }
    __syncthreads();
#endif

#pragma unroll
    for (int ktl = 0; ktl < 2; ++ktl) {           // two 32-K tiles per chunk
      const int k_tile = kc * 2 + ktl;
      // B fragments for both N-tiles (L2-resident, coalesced b128 pairs)
      union { uint4 u[2]; v16bf v; } b0, b1;
      {
        const unsigned short* p0 = Bp + ((size_t)nt0 * 32 + k_tile) * 512 + (size_t)lane * 16;
        const unsigned short* p1 = Bp + ((size_t)nt1 * 32 + k_tile) * 512 + (size_t)lane * 16;
        b0.u[0] = *(const uint4*)(p0);  b0.u[1] = *(const uint4*)(p0 + 8);
        b1.u[0] = *(const uint4*)(p1);  b1.u[1] = *(const uint4*)(p1 + 8);
      }
      // preload ALL 8 A fragments first so ds_loads batch ahead of the WMMA chain
      const int klocal = ktl * 32 + half * 8;
      union { uint4 u[2]; v16bf v; } af[8];
#pragma unroll
      for (int m = 0; m < 8; ++m) {
        const unsigned short* aptr = &sA[buf][(m * 16 + lane16) * CSTR + klocal];
        af[m].u[0] = *(const uint4*)(aptr);
        af[m].u[1] = *(const uint4*)(aptr + 16);
      }
#pragma unroll
      for (int m = 0; m < 8; ++m) {
        acc0[m] = __builtin_amdgcn_wmma_f32_16x16x32_bf16(
            false, af[m].v, false, b0.v, (short)0, acc0[m], false, false);
        acc1[m] = __builtin_amdgcn_wmma_f32_16x16x32_bf16(
            false, af[m].v, false, b1.v, (short)0, acc1[m], false, false);
      }
    }

#ifdef USE_ASYNC_LDS
    if (kc < 15) { wait_async0(); __syncthreads(); }
#endif
  }

  // store D (lane holds one column, 8 rows per tile)
  if (nt0 < KNT) {
    const int col = nt0 * 16 + lane16;
#pragma unroll
    for (int m = 0; m < 8; ++m)
#pragma unroll
      for (int e = 0; e < 8; ++e)
        logits[(size_t)(m * 16 + e + half * 8) * KVP + col] = acc0[m][e];
  }
  if (nt1 < KNT) {
    const int col = nt1 * 16 + lane16;
#pragma unroll
    for (int m = 0; m < 8; ++m)
#pragma unroll
      for (int e = 0; e < 8; ++e)
        logits[(size_t)(m * 16 + e + half * 8) * KVP + col] = acc1[m][e];
  }
}

// ---------------- per-row logsumexp ----------------
__global__ void row_lse_kernel(const float* __restrict__ logits, float* __restrict__ lse) {
  __shared__ float red[256];
  int row = blockIdx.x;
  const float* L = logits + (size_t)row * KVP;
  float mx = -1e30f;
  for (int c = threadIdx.x; c < KVOCAB; c += 256) mx = fmaxf(mx, L[c]);
  red[threadIdx.x] = mx; __syncthreads();
  for (int s = 128; s > 0; s >>= 1) {
    if (threadIdx.x < s) red[threadIdx.x] = fmaxf(red[threadIdx.x], red[threadIdx.x + s]);
    __syncthreads();
  }
  mx = red[0]; __syncthreads();
  float sum = 0.f;
  for (int c = threadIdx.x; c < KVOCAB; c += 256) sum += __expf(L[c] - mx);
  red[threadIdx.x] = sum; __syncthreads();
  for (int s = 128; s > 0; s >>= 1) {
    if (threadIdx.x < s) red[threadIdx.x] += red[threadIdx.x + s];
    __syncthreads();
  }
  if (threadIdx.x == 0) lse[row] = mx + __logf(red[0]);
}

// ---------------- per-batch top-8 of cum = (logit - lse) + prev_sc ----------------
__global__ void topk8_kernel(const float* __restrict__ logits, const float* __restrict__ lse,
                             const float* __restrict__ sc, int step,
                             float* __restrict__ cand_sc, int* __restrict__ cand_beam,
                             int* __restrict__ cand_tok) {
  __shared__ float sv[256][8];
  __shared__ int   si[256][8];
  const int b   = blockIdx.x;
  const int tid = threadIdx.x;

  float tv[8]; int tix[8];
#pragma unroll
  for (int k = 0; k < 8; ++k) { tv[k] = KNEG - 1.0f; tix[k] = 0; }

  for (int i = tid; i < KBEAM * KVOCAB; i += 256) {
    int beam = i / KVOCAB;
    int v    = i - beam * KVOCAB;
    int row  = b * KBEAM + beam;
    float lp  = logits[(size_t)row * KVP + v] - lse[row];
    float psc = (step == 0) ? 0.0f : sc[row * KSLEN + (step - 1)];
    float cum = lp + psc;
    if (v == TPAD) cum = KNEG;
    if (step == 0 && beam > 0) cum = KNEG;
    if (cum > tv[7]) {
      int k = 7;
      while (k > 0 && cum > tv[k - 1]) { tv[k] = tv[k - 1]; tix[k] = tix[k - 1]; --k; }
      tv[k] = cum; tix[k] = i;
    }
  }
#pragma unroll
  for (int k = 0; k < 8; ++k) { sv[tid][k] = tv[k]; si[tid][k] = tix[k]; }
  __syncthreads();

  for (int s = 128; s > 0; s >>= 1) {
    if (tid < s) {
      float av[8], bv[8], mv[8]; int ai[8], bi[8], mi[8];
#pragma unroll
      for (int k = 0; k < 8; ++k) {
        av[k] = sv[tid][k];     ai[k] = si[tid][k];
        bv[k] = sv[tid + s][k]; bi[k] = si[tid + s][k];
      }
      int ia = 0, ib = 0;
#pragma unroll
      for (int k = 0; k < 8; ++k) {
        bool takeA = (ib >= 8) || (ia < 8 && av[ia] >= bv[ib]);
        if (takeA) { mv[k] = av[ia]; mi[k] = ai[ia]; ++ia; }
        else       { mv[k] = bv[ib]; mi[k] = bi[ib]; ++ib; }
      }
#pragma unroll
      for (int k = 0; k < 8; ++k) { sv[tid][k] = mv[k]; si[tid][k] = mi[k]; }
    }
    __syncthreads();
  }
  if (tid < 8) {
    int f = si[0][tid];
    cand_sc[b * 8 + tid]   = sv[0][tid];
    cand_beam[b * 8 + tid] = f / KVOCAB;
    cand_tok[b * 8 + tid]  = f % KVOCAB;
  }
}

// ---------------- per-batch beam bookkeeping (serial, tiny) ----------------
__global__ void beam_update_kernel(const float* __restrict__ cand_sc,
                                   const int* __restrict__ cand_beam,
                                   const int* __restrict__ cand_tok,
                                   const int* __restrict__ tok_src, int* __restrict__ tok_dst,
                                   const float* __restrict__ sc_src, float* __restrict__ sc_dst,
                                   const float* __restrict__ fsc_src, float* __restrict__ fsc_dst,
                                   const int* __restrict__ ftok_src, int* __restrict__ ftok_dst,
                                   int step) {
  int b = blockIdx.x * blockDim.x + threadIdx.x;
  if (b >= KBSZ) return;
  const float* csc = cand_sc   + b * 8;
  const int*   cbm = cand_beam + b * 8;
  const int*   ctk = cand_tok  + b * 8;

  float asc[8]; int asrc[8];
  for (int j = 0; j < 4; ++j) { asc[j] = fsc_src[b * 4 + j]; asrc[j] = j; }
  for (int j = 0; j < 4; ++j) {
    bool iseos = (ctk[j] == TEOS);
    asc[4 + j]  = iseos ? csc[j] / (float)(step + 1) : KNEG;
    asrc[4 + j] = 4 + j;
  }
  for (int r = 0; r < 4; ++r) {
    int best = r;
    for (int j = r + 1; j < 8; ++j) if (asc[j] > asc[best]) best = j;
    float tv = asc[r]; asc[r] = asc[best]; asc[best] = tv;
    int   ts = asrc[r]; asrc[r] = asrc[best]; asrc[best] = ts;
  }
  for (int r = 0; r < 4; ++r) {
    fsc_dst[b * 4 + r] = asc[r];
    int s = asrc[r];
    int* dst = ftok_dst + ((size_t)b * 4 + r) * KL;
    if (s < 4) {
      const int* src = ftok_src + ((size_t)b * 4 + s) * KL;
      for (int l = 0; l < KL; ++l) dst[l] = src[l];
    } else {
      int j = s - 4;
      const int* src = tok_src + ((size_t)(b * 4 + cbm[j])) * KL;
      for (int l = 0; l < KL; ++l) dst[l] = (l == step + 1) ? TEOS : src[l];
    }
  }

  int act[4]; int na = 0;
  for (int j = 0; j < 8 && na < 4; ++j) if (ctk[j] != TEOS) act[na++] = j;
  for (int j = 0; j < 8 && na < 4; ++j) if (ctk[j] == TEOS) act[na++] = j;

  for (int r = 0; r < 4; ++r) {
    int j = act[r];
    int srow = b * 4 + cbm[j];
    int drow = b * 4 + r;
    for (int l = 0; l < KL; ++l) {
      int t = tok_src[(size_t)srow * KL + l];
      tok_dst[(size_t)drow * KL + l] = (l == step + 1) ? ctk[j] : t;
    }
    for (int l = 0; l < KSLEN; ++l) {
      float s = sc_src[(size_t)srow * KSLEN + l];
      sc_dst[(size_t)drow * KSLEN + l] = (l == step) ? csc[j] : s;
    }
  }
}

// ---------------- final: force EOS at step==MAXLEN, write outputs ----------------
__global__ void finalize_kernel(const float* __restrict__ logits, const float* __restrict__ lse,
                                const float* __restrict__ sc_cur, const int* __restrict__ tok_cur,
                                const float* __restrict__ fsc, const int* __restrict__ ftok,
                                int* __restrict__ out_tok, float* __restrict__ out_sc) {
  int b = blockIdx.x * blockDim.x + threadIdx.x;
  if (b >= KBSZ) return;
  float asc[8]; int asrc[8];
  for (int j = 0; j < 4; ++j) { asc[j] = fsc[b * 4 + j]; asrc[j] = j; }
  for (int j = 0; j < 4; ++j) {
    int row = b * 4 + j;
    float lpe = logits[(size_t)row * KVP + TEOS] - lse[row];
    asc[4 + j]  = (sc_cur[row * KSLEN + (KMAXLEN - 1)] + lpe) / (float)(KMAXLEN + 1);
    asrc[4 + j] = 4 + j;
  }
  for (int r = 0; r < 4; ++r) {
    int best = r;
    for (int j = r + 1; j < 8; ++j) if (asc[j] > asc[best]) best = j;
    float tv = asc[r]; asc[r] = asc[best]; asc[best] = tv;
    int   ts = asrc[r]; asrc[r] = asrc[best]; asrc[best] = ts;
  }
  for (int r = 0; r < 4; ++r) {
    out_sc[b * 4 + r] = asc[r];
    int s = asrc[r];
    int* dst = out_tok + ((size_t)b * 4 + r) * KL;
    if (s < 4) {
      const int* src = ftok + ((size_t)b * 4 + s) * KL;
      for (int l = 0; l < KL; ++l) dst[l] = src[l];
    } else {
      const int* src = tok_cur + ((size_t)(b * 4 + (s - 4))) * KL;
      for (int l = 0; l < KL; ++l) dst[l] = (l == KMAXLEN + 1) ? TEOS : src[l];
    }
  }
}

// ---------------- host driver ----------------
extern "C" void kernel_launch(void* const* d_in, const int* in_sizes, int n_in,
                              void* d_out, int out_size, void* d_ws, size_t ws_size,
                              hipStream_t stream) {
  const float* emb = (const float*)d_in[0];   // [VOCAB, D]
  const float* W   = (const float*)d_in[1];   // [D, VOCAB]
  const float* pos = (const float*)d_in[2];   // [SLEN, D]

  char* ws = (char*)d_ws;
  size_t off = 0;
  auto alloc = [&](size_t bytes) { size_t o = off; off = (o + bytes + 255) & ~(size_t)255; return o; };

  unsigned short* Bp = (unsigned short*)(ws + alloc((size_t)KNTP * 32 * 512 * 2));
  unsigned short* hA = (unsigned short*)(ws + alloc((size_t)KBB * KD * 2));
  float* logits      = (float*)(ws + alloc((size_t)KBB * KVP * 4));
  float* lse         = (float*)(ws + alloc((size_t)KBB * 4));
  int*   tokbuf[2]; tokbuf[0] = (int*)(ws + alloc((size_t)KBB * KL * 4));
                    tokbuf[1] = (int*)(ws + alloc((size_t)KBB * KL * 4));
  float* scbuf[2];  scbuf[0]  = (float*)(ws + alloc((size_t)KBB * KSLEN * 4));
                    scbuf[1]  = (float*)(ws + alloc((size_t)KBB * KSLEN * 4));
  float* fscbuf[2]; fscbuf[0] = (float*)(ws + alloc((size_t)KBSZ * KBEAM * 4));
                    fscbuf[1] = (float*)(ws + alloc((size_t)KBSZ * KBEAM * 4));
  int*   ftokbuf[2]; ftokbuf[0] = (int*)(ws + alloc((size_t)KBSZ * KBEAM * KL * 4));
                     ftokbuf[1] = (int*)(ws + alloc((size_t)KBSZ * KBEAM * KL * 4));
  float* cand_sc   = (float*)(ws + alloc((size_t)KBSZ * 8 * 4));
  int*   cand_beam = (int*)(ws + alloc((size_t)KBSZ * 8 * 4));
  int*   cand_tok  = (int*)(ws + alloc((size_t)KBSZ * 8 * 4));

  {
    size_t nfl = (size_t)KNTP * 32 * 32;
    pack_W_kernel<<<(unsigned)((nfl + 255) / 256), 256, 0, stream>>>(W, Bp);
  }
  init_state_kernel<<<(KBB * KL + 255) / 256, 256, 0, stream>>>(
      tokbuf[0], scbuf[0], fscbuf[0], ftokbuf[0]);

  for (int step = 0; step < KMAXLEN; ++step) {
    int cur = step & 1, nxt = cur ^ 1;
    build_h_kernel<<<KBB, 256, 0, stream>>>(emb, pos, tokbuf[cur], step, hA);
    gemm_logits_kernel<<<KNTP / 8, 128, 0, stream>>>(hA, Bp, logits);
    row_lse_kernel<<<KBB, 256, 0, stream>>>(logits, lse);
    topk8_kernel<<<KBSZ, 256, 0, stream>>>(logits, lse, scbuf[cur], step,
                                           cand_sc, cand_beam, cand_tok);
    beam_update_kernel<<<1, KBSZ, 0, stream>>>(cand_sc, cand_beam, cand_tok,
                                               tokbuf[cur], tokbuf[nxt],
                                               scbuf[cur], scbuf[nxt],
                                               fscbuf[cur], fscbuf[nxt],
                                               ftokbuf[cur], ftokbuf[nxt], step);
  }

  // after 64 steps the "current" buffers are index 0 again
  build_h_kernel<<<KBB, 256, 0, stream>>>(emb, pos, tokbuf[0], KMAXLEN, hA);
  gemm_logits_kernel<<<KNTP / 8, 128, 0, stream>>>(hA, Bp, logits);
  row_lse_kernel<<<KBB, 256, 0, stream>>>(logits, lse);

  int*   out_tok = (int*)d_out;                                  // [32,4,66] int32 bits
  float* out_sc  = (float*)d_out + (size_t)KBSZ * KBEAM * KL;    // [32,4] f32
  finalize_kernel<<<1, KBSZ, 0, stream>>>(logits, lse, scbuf[0], tokbuf[0],
                                          fscbuf[0], ftokbuf[0], out_tok, out_sc);
}
